// PixelMamba_74629351735622
// MI455X (gfx1250) — compile-verified, matrix operand order checked by reading, not run on hardware
//
#include <hip/hip_runtime.h>
#include <cmath>

// ---------------- problem constants ----------------
constexpr int BB   = 2;
constexpr int L    = 8192;
constexpr int DM   = 128;       // D_MODEL
constexpr int DIP  = 1552;      // D_IN_PROJ
constexpr int DI   = 512;       // D_INNER
constexpr int DXBC = 1024;      // D_XBC
constexpr int DS   = 256;       // D_STATE
constexpr int NH   = 16;        // NHEADS
constexpr int HD   = 32;        // HEADDIM
constexpr int CH   = 128;       // CHUNK
constexpr int NC   = 64;        // L / CHUNK
constexpr int ROWS = BB * L;    // 16384

// ---------------- WMMA plumbing (gfx1250, wave32) ----------------
typedef __attribute__((ext_vector_type(16))) _Float16 v16h;
typedef __attribute__((ext_vector_type(8)))  _Float16 v8h;
typedef __attribute__((ext_vector_type(8)))  float    v8f;
typedef unsigned int u32;
typedef __attribute__((ext_vector_type(4))) u32 v4u;
typedef __attribute__((ext_vector_type(8))) int v8i;
typedef __attribute__((ext_vector_type(4))) int v4i;

union BFrag { v16h v; v8h q[2]; _Float16 h[16]; };
union CFrag { v8f  v; float f[8]; };
union H8    { v8h  v; _Float16 h[8]; };
union F8    { v8f  v; float f[8]; };

__device__ __forceinline__ v8f wmma_f16(v16h a, v16h b, v8f c) {
  return __builtin_amdgcn_wmma_f32_16x16x32_f16(false, a, false, b,
                                                (short)0, c, false, false);
}
// K index held by (lane, element) in a 16-bit A/B fragment.
__device__ __forceinline__ int kmap(int lane, int e) {
  return ((lane >> 4) << 3) + ((e >> 3) << 4) + (e & 7);
}
// Fragment whose K-run is contiguous at base (16B aligned):
// elements 0..7 at [hl*8], elements 8..15 at [hl*8+16].
__device__ __forceinline__ v16h load_frag_contig(const _Float16* base, int hl) {
  BFrag f;
  const v8h* p = (const v8h*)(base + hl * 8);
  f.q[0] = p[0];
  f.q[1] = p[2];
  return f.v;
}
// fast transcendentals (hardware v_exp/v_log/v_rcp/v_rsq)
__device__ __forceinline__ float fexp(float x)  { return __expf(x); }
__device__ __forceinline__ float frcp(float x)  { return __builtin_amdgcn_rcpf(x); }
__device__ __forceinline__ float fsilu(float x) { return x * frcp(1.f + __expf(-x)); }

// ---------------- kernel 0: u = concat(x1,x2) @ fc0_w + fc0_b ----------------
__global__ void k_u(const float* __restrict__ x, const float* __restrict__ w,
                    const float* __restrict__ bias, _Float16* __restrict__ u_h) {
  int row = blockIdx.x, d = threadIdx.x;
  int b = row >> 13, l = row & 8191;
  int hh, ww;
  if (l < 4096) { hh = l >> 6; ww = l & 63; }
  else          { int l2 = l - 4096; hh = 63 - (l2 >> 6); ww = 63 - (l2 & 63); }
  float acc = bias[d];
#pragma unroll
  for (int c = 0; c < 3; ++c)
    acc += x[((b * 3 + c) * 64 + hh) * 64 + ww] * w[c * DM + d];
  u_h[row * DM + d] = (_Float16)acc;
}

// ---------------- transposed f16 weight converts ----------------
__global__ void k_cvt_ipw(const float* __restrict__ w, _Float16* __restrict__ wT) {
  int i = blockIdx.x * 256 + threadIdx.x;          // 1552*128
  int n = i >> 7, k = i & 127;
  wT[i] = (_Float16)w[k * DIP + n];
}
__global__ void k_cvt_opw(const float* __restrict__ w, _Float16* __restrict__ wT) {
  int i = blockIdx.x * 256 + threadIdx.x;          // 128*512
  int n = i >> 9, k = i & 511;
  wT[i] = (_Float16)w[k * DM + n];
}

// ---------------- kernel 2: in_proj GEMM, TDM-staged weights ----------------
// block = 256 thr (8 waves): 8 m-tiles x 1 n-tile. grid = 128*97.
__global__ void k_inproj(const _Float16* __restrict__ u_h,
                         const _Float16* __restrict__ wT,
                         const float* __restrict__ dt_bias,
                         float* __restrict__ z, float* __restrict__ xbc,
                         float* __restrict__ dt) {
  __shared__ __align__(16) _Float16 sW[16 * 128];  // [n_local][k]
  int nt = blockIdx.x % 97;
  int mg = blockIdx.x / 97;
  int wave = threadIdx.x >> 5, lane = threadIdx.x & 31;
  int lm = lane & 15, hl = lane >> 4;

  // --- Tensor Data Mover: DMA the 4KB weight panel (2048 f16) into LDS ---
  if (threadIdx.x < 32) {
    unsigned lds_addr = (unsigned)(size_t)(void*)sW;
    unsigned long long ga =
        (unsigned long long)(size_t)(const void*)(wT + nt * 2048);
    v4u g0 = { 1u,                                    // count=1, no gather
               lds_addr,                              // LDS dest
               (u32)ga,                               // global addr [31:0]
               (u32)((ga >> 32) & 0x01FFFFFFu) | (2u << 30) }; // [56:32] | type=2
    // group1: data_size=2B; tensor_dim0=198656; tensor_dim1=1;
    // tile_dim0=2048; tile_dim1=1; dim0_stride=198656
    v8i g1 = { (int)0x00010000,
               (int)0x08000000,      // tensor_dim0[15:0]=0x0800 << 16
               (int)0x00010003,      // tensor_dim0[31:16]=3 | tensor_dim1 lo=1
               (int)0x08000000,      // tile_dim0=0x0800 << 16
               1,                    // tile_dim1=1
               198656,               // tensor_dim0_stride lo32
               0, 0 };
    v4i gz = { 0, 0, 0, 0 };
#if __clang_major__ >= 23
    v8i gz8 = { 0, 0, 0, 0, 0, 0, 0, 0 };
    __builtin_amdgcn_tensor_load_to_lds(g0, g1, gz, gz, gz8, 0);
#else
    __builtin_amdgcn_tensor_load_to_lds(g0, g1, gz, gz, 0);
#endif
    __builtin_amdgcn_s_wait_tensorcnt(0);
  }
  __syncthreads();

  int mt = mg * 8 + wave;
  int m = mt * 16 + lm;
  CFrag acc;
#pragma unroll
  for (int r = 0; r < 8; ++r) acc.f[r] = 0.f;
#pragma unroll
  for (int ks = 0; ks < 4; ++ks) {
    v16h a = load_frag_contig(u_h + m * DM + ks * 32, hl);
    v16h b = load_frag_contig(sW + lm * 128 + ks * 32, hl);
    acc.v = wmma_f16(a, b, acc.v);
  }
  int n = nt * 16 + lm;
#pragma unroll
  for (int r = 0; r < 8; ++r) {
    int mm = mt * 16 + hl * 8 + r;
    float val = acc.f[r];
    if (n < DI)              z[mm * DI + n] = val;
    else if (n < DI + DXBC)  xbc[mm * DXBC + (n - DI)] = val;
    else {
      int h = n - (DI + DXBC);
      float xv = val + dt_bias[h];
      dt[mm * NH + h] = (xv > 20.f) ? xv : __logf(1.f + fexp(xv));  // softplus
    }
  }
}

// ---------------- kernel 3: conv4 + SiLU; emit row and chunk-transposed -----
__global__ void k_conv(const float* __restrict__ xbc, const float* __restrict__ cw,
                       const float* __restrict__ cb,
                       _Float16* __restrict__ xs_h, _Float16* __restrict__ B_h,
                       _Float16* __restrict__ C_h, _Float16* __restrict__ xsT,
                       _Float16* __restrict__ BT) {
  int gid = blockIdx.x * blockDim.x + threadIdx.x;
  int ch  = gid & 1023;
  int row = gid >> 10;
  int b = row >> 13, l = row & 8191;
  float acc = cb[ch];
#pragma unroll
  for (int j = 0; j < 4; ++j) {
    int li = l - 3 + j;
    if (li >= 0) acc += xbc[((b << 13) + li) * DXBC + ch] * cw[ch * 4 + j];
  }
  _Float16 hv = (_Float16)fsilu(acc);
  int bc = row >> 7, t = row & 127;
  if (ch < DI) {
    xs_h[row * DI + ch] = hv;
    int h = ch >> 5, p = ch & 31;
    xsT[((bc * NH + h) * HD + p) * CH + t] = hv;          // [bc][h][p][t]
  } else if (ch < DI + DS) {
    int n = ch - DI;
    B_h[row * DS + n] = hv;
    BT[(bc * DS + n) * CH + t] = hv;                       // [bc][n][t]
  } else {
    C_h[row * DS + (ch - DI - DS)] = hv;
  }
}

// ---------------- kernel 4: cumsum of a = dt*A within chunk -----------------
__global__ void k_cum(const float* __restrict__ dt, const float* __restrict__ A_log,
                      float* __restrict__ cum) {
  int bc = blockIdx.x / NH, h = blockIdx.x % NH;
  int t = threadIdx.x;
  float A = -fexp(A_log[h]);
  __shared__ float s[CH];
  s[t] = dt[(bc * CH + t) * NH + h] * A;
  __syncthreads();
  for (int off = 1; off < CH; off <<= 1) {
    float v = (t >= off) ? s[t - off] : 0.f;
    __syncthreads();
    s[t] += v;
    __syncthreads();
  }
  cum[blockIdx.x * CH + t] = s[t];
}

// ---------------- kernel 5: per-chunk CB + masked attn @ x ------------------
__global__ void k_intra(const _Float16* __restrict__ B_h,
                        const _Float16* __restrict__ C_h,
                        const _Float16* __restrict__ xsT,
                        const float* __restrict__ cum,
                        const float* __restrict__ dt,
                        float* __restrict__ y) {
  __shared__ _Float16 sCB[CH * CH];   // 32 KB
  __shared__ float sCum[NH * CH];     // 8 KB  [h][t]
  __shared__ float sDt[CH * NH];      // 8 KB  [t][h]
  int bc = blockIdx.x;
  int rowbase = bc * CH;
  int tid = threadIdx.x;
  int wave = tid >> 5, lane = tid & 31, lm = lane & 15, hl = lane >> 4;
  for (int i = tid; i < NH * CH; i += 512) {
    sCum[i] = cum[bc * NH * CH + i];
    sDt[i]  = dt[rowbase * NH + i];
  }
  __syncthreads();
  // Phase 1: CB = C_r(128x256) @ B_r^T, lower-triangular tiles
  for (int tile = wave; tile < 64; tile += 16) {
    int ti = tile >> 3, sj = tile & 7;
    if (sj > ti) continue;
    CFrag acc;
#pragma unroll
    for (int r = 0; r < 8; ++r) acc.f[r] = 0.f;
#pragma unroll
    for (int ks = 0; ks < 8; ++ks) {
      v16h a = load_frag_contig(C_h + (rowbase + ti * 16 + lm) * DS + ks * 32, hl);
      v16h b = load_frag_contig(B_h + (rowbase + sj * 16 + lm) * DS + ks * 32, hl);
      acc.v = wmma_f16(a, b, acc.v);
    }
#pragma unroll
    for (int r = 0; r < 8; ++r)
      sCB[(ti * 16 + hl * 8 + r) * CH + sj * 16 + lm] = (_Float16)acc.f[r];
  }
  __syncthreads();
  // Phase 2: wave == head; y_intra = (CB .* decay .* dt) @ x
  int h = wave;
  int xbase = (bc * NH + h) * HD * CH;
  for (int ot = 0; ot < 16; ++ot) {
    int ti = ot >> 1, pj = ot & 1;
    CFrag acc;
#pragma unroll
    for (int r = 0; r < 8; ++r) acc.f[r] = 0.f;
    for (int ks4 = 0; ks4 <= (ti >> 1); ++ks4) {   // causal K blocks only
      BFrag a;
#pragma unroll
      for (int e = 0; e < 16; ++e) {
        int t = ti * 16 + lm;
        int s = ks4 * 32 + kmap(lane, e);
        float v = 0.f;
        if (s <= t)
          v = (float)sCB[t * CH + s] *
              fexp(sCum[h * CH + t] - sCum[h * CH + s]) * sDt[s * NH + h];
        a.h[e] = (_Float16)v;
      }
      v16h b = load_frag_contig(xsT + xbase + (pj * 16 + lm) * CH + ks4 * 32, hl);
      acc.v = wmma_f16(a.v, b, acc.v);
    }
#pragma unroll
    for (int r = 0; r < 8; ++r) {
      int t = ti * 16 + hl * 8 + r;
      y[(rowbase + t) * DI + h * HD + pj * 16 + lm] = acc.f[r];
    }
  }
}

// ---------------- kernel 6: chunk states = (w.*x)^T @ B ---------------------
__global__ void k_states(const _Float16* __restrict__ BT,
                         const _Float16* __restrict__ xsT,
                         const float* __restrict__ cum,
                         const float* __restrict__ dt,
                         float* __restrict__ states) {
  int bch = blockIdx.x;
  int bc = bch / NH, h = bch % NH;
  int rowbase = bc * CH;
  __shared__ float w[CH];
  int tid = threadIdx.x;
  float clast = cum[bch * CH + CH - 1];
  w[tid] = fexp(clast - cum[bch * CH + tid]) * dt[(rowbase + tid) * NH + h];
  __syncthreads();
  int wave = tid >> 5, lane = tid & 31, lm = lane & 15, hl = lane >> 4;
  int xrow = (bc * NH + h) * HD;
  for (int tile = wave; tile < 32; tile += 4) {
    int pt = tile >> 4, nt = tile & 15;
    CFrag acc;
#pragma unroll
    for (int r = 0; r < 8; ++r) acc.f[r] = 0.f;
#pragma unroll
    for (int s4 = 0; s4 < 4; ++s4) {
      const v8h* px = (const v8h*)(xsT + (xrow + pt * 16 + lm) * CH + s4 * 32 + hl * 8);
      H8 x0, x1; x0.v = px[0]; x1.v = px[2];
      BFrag a;
#pragma unroll
      for (int e = 0; e < 8; ++e) {
        a.h[e]     = (_Float16)(w[s4 * 32 + hl * 8 + e]      * (float)x0.h[e]);
        a.h[8 + e] = (_Float16)(w[s4 * 32 + 16 + hl * 8 + e] * (float)x1.h[e]);
      }
      v16h b = load_frag_contig(BT + (bc * DS + nt * 16 + lm) * CH + s4 * 32, hl);
      acc.v = wmma_f16(a.v, b, acc.v);
    }
#pragma unroll
    for (int r = 0; r < 8; ++r)
      states[bch * HD * DS + (pt * 16 + hl * 8 + r) * DS + nt * 16 + lm] = acc.f[r];
  }
}

// ---------------- kernel 7: sequential inter-chunk scan (in place) ----------
__global__ void k_scan(const float* __restrict__ cum, float* __restrict__ states) {
  int b = blockIdx.x / NH, h = blockIdx.x % NH;
  int tid = threadIdx.x;
  float S[32];
#pragma unroll
  for (int j = 0; j < 32; ++j) S[j] = 0.f;
  for (int c = 0; c < NC; ++c) {
    int bch = (b * NC + c) * NH + h;
    float cd = fexp(cum[bch * CH + CH - 1]);
    int base = bch * HD * DS;
#pragma unroll
    for (int j = 0; j < 32; ++j) {
      int idx = base + j * 256 + tid;
      float st = states[idx];
      states[idx] = S[j];
      S[j] = S[j] * cd + st;
    }
  }
}

// ---------------- kernel 8: y_inter = exp(cum) .* (C @ prev^T) --------------
__global__ void k_inter(const _Float16* __restrict__ C_h,
                        const float* __restrict__ prev,
                        const float* __restrict__ cum,
                        float* __restrict__ y) {
  int bch = blockIdx.x;
  int bc = bch / NH, h = bch % NH;
  int rowbase = bc * CH;
  int tid = threadIdx.x;
  int wave = tid >> 5, lane = tid & 31, lm = lane & 15, hl = lane >> 4;
  int pbase = bch * HD * DS;
  for (int tile = wave; tile < 16; tile += 4) {
    int ti = tile >> 1, pj = tile & 1;
    CFrag acc;
#pragma unroll
    for (int r = 0; r < 8; ++r) acc.f[r] = 0.f;
#pragma unroll
    for (int nk = 0; nk < 8; ++nk) {
      v16h a = load_frag_contig(C_h + (rowbase + ti * 16 + lm) * DS + nk * 32, hl);
      const v8f* pp = (const v8f*)(prev + pbase + (pj * 16 + lm) * DS + nk * 32 + hl * 8);
      F8 b0, b1; b0.v = pp[0]; b1.v = pp[2];
      BFrag b;
#pragma unroll
      for (int e = 0; e < 8; ++e) {
        b.h[e]     = (_Float16)b0.f[e];
        b.h[8 + e] = (_Float16)b1.f[e];
      }
      acc.v = wmma_f16(a, b.v, acc.v);
    }
#pragma unroll
    for (int r = 0; r < 8; ++r) {
      int t = ti * 16 + hl * 8 + r;
      y[(rowbase + t) * DI + h * HD + pj * 16 + lm] += acc.f[r] * fexp(cum[bch * CH + t]);
    }
  }
}

// ---------------- kernel 9a: +D*xs, silu(z) gate, RMSNorm -> f16 ------------
__global__ void k_gate(const float* __restrict__ y, const _Float16* __restrict__ xs_h,
                       const float* __restrict__ z, const float* __restrict__ Dp,
                       const float* __restrict__ norm_w, _Float16* __restrict__ yn_h) {
  __shared__ float red[DI];
  int row = blockIdx.x, ch = threadIdx.x;
  float v = y[row * DI + ch] + Dp[ch >> 5] * (float)xs_h[row * DI + ch];
  v *= fsilu(z[row * DI + ch]);
  red[ch] = v * v;
  __syncthreads();
  for (int off = DI / 2; off > 0; off >>= 1) {
    if (ch < off) red[ch] += red[ch + off];
    __syncthreads();
  }
  float scale = __builtin_amdgcn_rsqf(red[0] * (1.f / DI) + 1e-5f);
  yn_h[row * DI + ch] = (_Float16)(v * scale * norm_w[ch]);
}

// ---------------- kernel 9b: out_proj GEMM ----------------------------------
__global__ void k_outproj(const _Float16* __restrict__ yn_h,
                          const _Float16* __restrict__ wT,
                          float* __restrict__ out) {
  int wave = blockIdx.x * 8 + (threadIdx.x >> 5);
  int lane = threadIdx.x & 31, lm = lane & 15, hl = lane >> 4;
  int mt = wave >> 3, nt = wave & 7;
  CFrag acc;
#pragma unroll
  for (int r = 0; r < 8; ++r) acc.f[r] = 0.f;
#pragma unroll
  for (int ks = 0; ks < 16; ++ks) {
    v16h a = load_frag_contig(yn_h + (mt * 16 + lm) * DI + ks * 32, hl);
    v16h b = load_frag_contig(wT + (nt * 16 + lm) * DI + ks * 32, hl);
    acc.v = wmma_f16(a, b, acc.v);
  }
#pragma unroll
  for (int r = 0; r < 8; ++r)
    out[(mt * 16 + hl * 8 + r) * DM + nt * 16 + lm] = acc.f[r];
}

// ---------------- host launcher ----------------
extern "C" void kernel_launch(void* const* d_in, const int* in_sizes, int n_in,
                              void* d_out, int out_size, void* d_ws, size_t ws_size,
                              hipStream_t stream) {
  (void)in_sizes; (void)n_in; (void)out_size; (void)ws_size;
  const float* x          = (const float*)d_in[0];
  const float* fc0_w      = (const float*)d_in[1];
  const float* fc0_b      = (const float*)d_in[2];
  const float* in_proj_w  = (const float*)d_in[3];
  const float* conv_w     = (const float*)d_in[4];
  const float* conv_b     = (const float*)d_in[5];
  const float* dt_bias    = (const float*)d_in[6];
  const float* A_log      = (const float*)d_in[7];
  const float* Dp         = (const float*)d_in[8];
  const float* norm_w     = (const float*)d_in[9];
  const float* out_proj_w = (const float*)d_in[10];
  float* out = (float*)d_out;

  char* p = (char*)d_ws;
  auto alloc = [&](size_t bytes) -> char* {
    char* r = p; p += (bytes + 255) & ~size_t(255); return r;
  };
  _Float16* u_h   = (_Float16*)alloc((size_t)ROWS * DM * 2);
  _Float16* wT    = (_Float16*)alloc((size_t)DIP * DM * 2);     // in_proj^T [n][k]
  _Float16* opwT  = (_Float16*)alloc((size_t)DM * DI * 2);      // out_proj^T [n][k]
  float*    z     = (float*)   alloc((size_t)ROWS * DI * 4);
  float*    xbc   = (float*)   alloc((size_t)ROWS * DXBC * 4);
  float*    dt    = (float*)   alloc((size_t)ROWS * NH * 4);
  _Float16* xs_h  = (_Float16*)alloc((size_t)ROWS * DI * 2);
  _Float16* B_h   = (_Float16*)alloc((size_t)ROWS * DS * 2);
  _Float16* C_h   = (_Float16*)alloc((size_t)ROWS * DS * 2);
  _Float16* xsT   = (_Float16*)alloc((size_t)BB * NC * NH * HD * CH * 2);
  _Float16* BT    = (_Float16*)alloc((size_t)BB * NC * DS * CH * 2);
  float*    cum   = (float*)   alloc((size_t)BB * NC * NH * CH * 4);
  float*    y     = (float*)   alloc((size_t)ROWS * DI * 4);
  float*    states= (float*)   alloc((size_t)BB * NC * NH * HD * DS * 4);
  _Float16* yn_h  = (_Float16*)alloc((size_t)ROWS * DI * 2);

  k_u       <<<ROWS, DM, 0, stream>>>(x, fc0_w, fc0_b, u_h);
  k_cvt_ipw <<<(DIP * DM) / 256, 256, 0, stream>>>(in_proj_w, wT);
  k_cvt_opw <<<(DM * DI) / 256, 256, 0, stream>>>(out_proj_w, opwT);
  k_inproj  <<<128 * 97, 256, 0, stream>>>(u_h, wT, dt_bias, z, xbc, dt);
  k_conv    <<<(ROWS * DXBC) / 256, 256, 0, stream>>>(xbc, conv_w, conv_b,
                                                      xs_h, B_h, C_h, xsT, BT);
  k_cum     <<<BB * NC * NH, CH, 0, stream>>>(dt, A_log, cum);
  k_intra   <<<BB * NC, 512, 0, stream>>>(B_h, C_h, xsT, cum, dt, y);
  k_states  <<<BB * NC * NH, 128, 0, stream>>>(BT, xsT, cum, dt, states);
  k_scan    <<<BB * NH, 256, 0, stream>>>(cum, states);
  k_inter   <<<BB * NC * NH, 128, 0, stream>>>(C_h, states, cum, y);
  k_gate    <<<ROWS, DI, 0, stream>>>(y, xs_h, z, Dp, norm_w, yn_h);
  k_outproj <<<(1024 * 8) / 8, 256, 0, stream>>>(yn_h, opwT, out);
}
